// CrossLayerTranscoder_78735340470778
// MI455X (gfx1250) — matrix operand order repaired and checked by reading, not compiled, not run on hardware
//
#include <hip/hip_runtime.h>
#include <hip/hip_bf16.h>

// CrossLayerTranscoder on MI455X (gfx1250, wave32, WMMA).
// fp32 weights are streamed from HBM exactly once with non-temporal hints,
// converted in-register / in-LDS to f16, and fed to V_WMMA_F32_16X16X32_F16
// with fp32 accumulation. Encode writes f16 activations to d_ws (9.44 MB),
// which stay L2-resident for the decode phase. Decode double-buffers its
// LDS weight tile so staging overlaps the WMMA stream.

typedef _Float16 f16;
typedef __attribute__((ext_vector_type(16))) _Float16 v16h;
typedef __attribute__((ext_vector_type(8)))  _Float16 v8h;
typedef __attribute__((ext_vector_type(8)))  float    v8f;
typedef __attribute__((ext_vector_type(4)))  float    v4f;

#define N_LAYERS 6
#define D_T      6144
#define D_M      768
#define TOK      128
#define KCHUNK   1536              // decode K split (6144 / 4)
#define KSTEPS   (KCHUNK / 32)     // 48 inner iterations

// ---------------------------------------------------------------------------
// Encode: acts[l,t,f] = relu(x[l,t,:] . W_enc[l,f,:] + b_enc[l,f]), f16 out.
// Block = 8 waves (256 thr). Block tile: 128 tokens x 128 features.
// Wave w owns token rows [16w, 16w+16) x all 128 features (8 N-subtiles).
// ---------------------------------------------------------------------------
__global__ __launch_bounds__(256)
void xlt_encode(const float* __restrict__ x, const float* __restrict__ Wenc,
                const float* __restrict__ benc, f16* __restrict__ acts)
{
    const int l     = blockIdx.y;
    const int nbase = blockIdx.x * 128;
    const int wave  = threadIdx.x >> 5;
    const int lane  = threadIdx.x & 31;
    const int lo    = lane & 15;
    const int hi    = lane >> 4;          // 0 or 1 (lane half)
    const int tok   = wave * 16 + lo;

    v8f acc[8];
#pragma unroll
    for (int n = 0; n < 8; ++n) acc[n] = (v8f)0.0f;

    const float* xrow = x + ((size_t)l * TOK + tok) * D_M;

    for (int k0 = 0; k0 < D_M; k0 += 32) {
        // A fragment (16x32 f16): lanes 0-15 hold K 0..7 & 16..23 of row M=lo,
        // lanes 16-31 hold K 8..15 & 24..31. x is reused across blocks -> RT.
        const float* ap = xrow + k0 + hi * 8;
        v4f a0 = *(const v4f*)(ap);
        v4f a1 = *(const v4f*)(ap + 4);
        v4f a2 = *(const v4f*)(ap + 16);
        v4f a3 = *(const v4f*)(ap + 20);
        v16h afrag;
#pragma unroll
        for (int e = 0; e < 4; ++e) {
            afrag[e]      = (f16)a0[e];
            afrag[4 + e]  = (f16)a1[e];
            afrag[8 + e]  = (f16)a2[e];
            afrag[12 + e] = (f16)a3[e];
        }
#pragma unroll
        for (int n = 0; n < 8; ++n) {
            // B fragment (32x16 f16): column = feature f (contiguous K along
            // d_model in W_enc row f). Lanes 0-15: K 0..7 & 16..23, lanes 16-31: K 8..15 & 24..31.
            // W_enc is streamed exactly once -> non-temporal.
            const int f = nbase + n * 16 + lo;
            const v4f* bp = (const v4f*)(Wenc + ((size_t)l * D_T + f) * D_M
                                              + k0 + hi * 8);
            v4f b0 = __builtin_nontemporal_load(bp);
            v4f b1 = __builtin_nontemporal_load(bp + 1);
            v4f b2 = __builtin_nontemporal_load(bp + 4);
            v4f b3 = __builtin_nontemporal_load(bp + 5);
            v16h bfrag;
#pragma unroll
            for (int e = 0; e < 4; ++e) {
                bfrag[e]      = (f16)b0[e];
                bfrag[4 + e]  = (f16)b1[e];
                bfrag[8 + e]  = (f16)b2[e];
                bfrag[12 + e] = (f16)b3[e];
            }
            acc[n] = __builtin_amdgcn_wmma_f32_16x16x32_f16(
                false, afrag, false, bfrag, (short)0, acc[n], false, false);
        }
    }

    // Epilogue: bias + ReLU, store f16 acts [l][token][feature].
    // C/D layout: VGPR r, lane -> token = 16w + r + 8*hi, feature = nbase+16n+lo.
#pragma unroll
    for (int n = 0; n < 8; ++n) {
        const int f = nbase + n * 16 + lo;
        const float bias = benc[l * D_T + f];
#pragma unroll
        for (int r = 0; r < 8; ++r) {
            const int t = wave * 16 + r + hi * 8;
            float v = acc[n][r] + bias;
            v = v > 0.0f ? v : 0.0f;
            acts[((size_t)l * TOK + t) * D_T + f] = (f16)v;
        }
    }
}

// ---------------------------------------------------------------------------
// Init: out[j,t,d] = b_dec[j,d]
// ---------------------------------------------------------------------------
__global__ void xlt_init(const float* __restrict__ bdec, float* __restrict__ out)
{
    const int idx = blockIdx.x * blockDim.x + threadIdx.x;
    const int d = idx % D_M;
    const int j = idx / (D_M * TOK);
    out[idx] = bdec[j * D_M + d];
}

// ---------------------------------------------------------------------------
// Decode: out[j,t,d] += sum_f acts[i,t,f] * W_dec[i,j,f,d] for i <= j.
// Grid: (d-tile of 128, K-chunk of 1536, z = i*6+j; i>j blocks exit).
// W_dec is row-major [f][d] (K-major for B) -> stage through LDS transposed
// to [n][k] f16 so B fragments are contiguous 16B ds_load_b128 reads.
// LDS tile is double-buffered: tile s+1 is staged (NT global loads + ds
// stores) while tile s feeds 8 WMMAs. Partials accumulate via atomic f32.
// ---------------------------------------------------------------------------
__global__ __launch_bounds__(256)
void xlt_decode(const f16* __restrict__ acts, const float* __restrict__ Wdec,
                float* __restrict__ out)
{
    const int z = blockIdx.z;
    const int i = z / N_LAYERS;
    const int j = z % N_LAYERS;
    if (i > j) return;                    // uniform per-block, safe vs barriers

    const int nbase  = blockIdx.x * 128;
    const int kstart = blockIdx.y * KCHUNK;

    __shared__ __align__(16) f16 lb[2][128][40];  // [buf][n][k], 80B row pitch

    const int tid  = threadIdx.x;
    const int wave = tid >> 5;
    const int lane = tid & 31;
    const int lo   = lane & 15;
    const int hi   = lane >> 4;
    const int tok  = wave * 16 + lo;

    v8f acc[8];
#pragma unroll
    for (int n = 0; n < 8; ++n) acc[n] = (v8f)0.0f;

    const f16*   arow  = acts + ((size_t)i * TOK + tok) * D_T;
    const float* wbase = Wdec + ((size_t)(i * N_LAYERS + j) * D_T) * D_M + nbase;

    // Cooperative staging: 32 k-rows x 128 d-cols, 16 floats per thread.
    const int lr = tid >> 3;             // k-row 0..31
    const int lc = (tid & 7) * 16;       // d-col base

    // --- stage tile 0 into buffer 0 ---
    {
        const float* wp = wbase + (size_t)(kstart + lr) * D_M + lc;
#pragma unroll
        for (int u = 0; u < 4; ++u) {
            v4f w = __builtin_nontemporal_load((const v4f*)(wp + u * 4));
#pragma unroll
            for (int e = 0; e < 4; ++e)
                lb[0][lc + u * 4 + e][lr] = (f16)w[e];
        }
    }

    for (int s = 0; s < KSTEPS; ++s) {
        __syncthreads();                 // buffer s&1 is ready

        // Stage tile s+1 into the other buffer while computing tile s.
        if (s + 1 < KSTEPS) {
            const int cur = (s + 1) & 1;
            const float* wp = wbase + (size_t)(kstart + (s + 1) * 32 + lr) * D_M + lc;
            // keep the HBM stream ahead of the double buffer
            if (s + 2 < KSTEPS)
                __builtin_prefetch(wp + (size_t)32 * D_M, 0, 0);
#pragma unroll
            for (int u = 0; u < 4; ++u) {
                v4f w = __builtin_nontemporal_load((const v4f*)(wp + u * 4));
#pragma unroll
                for (int e = 0; e < 4; ++e)
                    lb[cur][lc + u * 4 + e][lr] = (f16)w[e];
            }
        }

        // A fragment from f16 acts (K = features), L2-resident.
        const int buf = s & 1;
        const f16* ap = arow + kstart + s * 32 + hi * 8;
        v8h alo = *(const v8h*)(ap);
        v8h ahi = *(const v8h*)(ap + 16);
        v16h afrag;
#pragma unroll
        for (int e = 0; e < 8; ++e) { afrag[e] = alo[e]; afrag[8 + e] = ahi[e]; }

#pragma unroll
        for (int n = 0; n < 8; ++n) {
            const int nn = n * 16 + lo;
            v8h blo = *(const v8h*)&lb[buf][nn][hi * 8];
            v8h bhi = *(const v8h*)&lb[buf][nn][hi * 8 + 16];
            v16h bfrag;
#pragma unroll
            for (int e = 0; e < 8; ++e) { bfrag[e] = blo[e]; bfrag[8 + e] = bhi[e]; }
            acc[n] = __builtin_amdgcn_wmma_f32_16x16x32_f16(
                false, afrag, false, bfrag, (short)0, acc[n], false, false);
        }
    }

#pragma unroll
    for (int n = 0; n < 8; ++n) {
        const int d = nbase + n * 16 + lo;
#pragma unroll
        for (int r = 0; r < 8; ++r) {
            const int t = wave * 16 + r + hi * 8;
            atomicAdd(&out[((size_t)j * TOK + t) * D_M + d], acc[n][r]);
        }
    }
}

// ---------------------------------------------------------------------------
extern "C" void kernel_launch(void* const* d_in, const int* in_sizes, int n_in,
                              void* d_out, int out_size, void* d_ws, size_t ws_size,
                              hipStream_t stream)
{
    const float* x    = (const float*)d_in[0];
    const float* Wenc = (const float*)d_in[1];
    const float* benc = (const float*)d_in[2];
    const float* bdec = (const float*)d_in[3];
    const float* Wdec = (const float*)d_in[4];
    // d_in[5] = dec_mask: triangle is known statically, unused.
    float* out  = (float*)d_out;
    f16*   acts = (f16*)d_ws;            // needs N_LAYERS*TOK*D_T*2 = 9.44 MB

    // 1) Encode -> f16 activations in workspace.
    xlt_encode<<<dim3(D_T / 128, N_LAYERS), 256, 0, stream>>>(x, Wenc, benc, acts);

    // 2) Seed output with decoder bias.
    xlt_init<<<(N_LAYERS * TOK * D_M) / 256, 256, 0, stream>>>(bdec, out);

    // 3) Cross-layer decode, atomically accumulated.
    xlt_decode<<<dim3(D_M / 128, D_T / KCHUNK, N_LAYERS * N_LAYERS), 256, 0, stream>>>(
        acts, Wdec, out);
}
